// AttentionHead_90838558310710
// MI455X (gfx1250) — matrix operand order repaired and checked by reading, not compiled, not run on hardware
//
#include <hip/hip_runtime.h>
#include <hip/hip_bf16.h>

// ---------------------------------------------------------------------------
// Single-head causal attention, B=4 T=4096 E=1024 H=64, fp32 in/out.
// bf16 WMMA (f32 accumulate) flash-attention pipeline for MI455X (gfx1250).
// K tiles staged through LDS with global_load_async_to_lds (ASYNCcnt
// double-buffering); conversions via v_cvt_pk_bf16_f32.
// ---------------------------------------------------------------------------

#define BDIM 4
#define TDIM 4096
#define EDIM 1024
#define HDIM 64

typedef __attribute__((ext_vector_type(16))) __bf16 v16bf;
typedef __attribute__((ext_vector_type(8)))  float  v8f;

union Frag {
    unsigned int u[8];   // 8 VGPRs of packed bf16 pairs
    v16bf        v;
};

// ---- f32x2 -> packed bf16x2 (single VALU op) ------------------------------
#if __has_builtin(__builtin_amdgcn_cvt_pk_bf16_f32)
__device__ __forceinline__ unsigned int pack_bf16x2(float lo, float hi) {
    auto r = __builtin_amdgcn_cvt_pk_bf16_f32(lo, hi);
    return __builtin_bit_cast(unsigned int, r);
}
#else
__device__ __forceinline__ unsigned int pack_bf16x2(float lo, float hi) {
    unsigned int r;
    asm("v_cvt_pk_bf16_f32 %0, %1, %2" : "=v"(r) : "v"(lo), "v"(hi));
    return r;
}
#endif

__device__ __forceinline__ unsigned short f32_to_bf16(float f) {
    return (unsigned short)(pack_bf16x2(f, f) & 0xFFFFu);
}

// A-matrix (16x32 bf16) K-offset for VGPR v: lane half hl holds K pairs
// {hl*16+2v, +1} (contiguous order, matching the B fragments).
__device__ __forceinline__ int a_koff(int v, int hl) {
    return hl * 16 + 2 * v;
}

// ---------------------------------------------------------------------------
// Kernel 0: Wq/Wk/Wv fp32 [E][H]  ->  bf16 transposed [3][H][E]
// ---------------------------------------------------------------------------
__global__ void wprep_kernel(const float* __restrict__ Wq,
                             const float* __restrict__ Wk,
                             const float* __restrict__ Wv,
                             unsigned short* __restrict__ Wt) {
    int idx = blockIdx.x * blockDim.x + threadIdx.x;      // over 3*H*E
    if (idx >= 3 * HDIM * EDIM) return;
    int mat = idx / (HDIM * EDIM);
    int r   = idx % (HDIM * EDIM);
    int h   = r / EDIM;
    int e   = r % EDIM;
    const float* W = (mat == 0) ? Wq : (mat == 1) ? Wk : Wv;
    Wt[idx] = f32_to_bf16(W[(size_t)e * HDIM + h]);
}

// ---------------------------------------------------------------------------
// Kernel 1: QKV projection.  4 waves/block, each wave = 16 rows x 64 cols of
// Q, K, V.  Q,K stored bf16 [B*T][H]; V stored bf16 transposed [B][H][T]
// (packed b128 stores: consecutive accumulator rows -> consecutive t).
// ---------------------------------------------------------------------------
__global__ __launch_bounds__(128) void qkv_kernel(
        const float* __restrict__ x,           // [B*T][E] f32
        const unsigned short* __restrict__ Wt, // [3][H][E] bf16
        unsigned short* __restrict__ Qw,       // [B*T][H] bf16
        unsigned short* __restrict__ Kw,       // [B*T][H] bf16
        unsigned short* __restrict__ Vtw) {    // [B][H][T] bf16
    const int lane = threadIdx.x & 31;
    const int wave = threadIdx.x >> 5;
    const int hl   = lane >> 4;
    const int n16  = lane & 15;
    const int rowBase = blockIdx.x * 64 + wave * 16;      // flat row

    v8f acc[3][4];
#pragma unroll
    for (int m = 0; m < 3; ++m)
#pragma unroll
        for (int nt = 0; nt < 4; ++nt) acc[m][nt] = v8f{};

    const float* xrow = x + (size_t)(rowBase + n16) * EDIM;

    for (int kc = 0; kc < EDIM; kc += 32) {
        // B fragments: 12 x 8 contiguous dwords (merge to b128 loads)
        Frag bfr[12];
#pragma unroll
        for (int mat = 0; mat < 3; ++mat)
#pragma unroll
            for (int nt = 0; nt < 4; ++nt) {
                const unsigned short* wp =
                    Wt + ((size_t)(mat * HDIM + nt * 16 + n16)) * EDIM + kc;
#pragma unroll
                for (int v = 0; v < 8; ++v)
                    bfr[mat * 4 + nt].u[v] =
                        *(const unsigned int*)(wp + hl * 16 + 2 * v);
            }
        // A fragment: x rows, K = kc..kc+31, converted to bf16
        Frag a;
#pragma unroll
        for (int v = 0; v < 8; ++v) {
            const float2 f = *(const float2*)(xrow + kc + a_koff(v, hl));
            a.u[v] = pack_bf16x2(f.x, f.y);
        }
#pragma unroll
        for (int mat = 0; mat < 3; ++mat)
#pragma unroll
            for (int nt = 0; nt < 4; ++nt)
                acc[mat][nt] = __builtin_amdgcn_wmma_f32_16x16x32_bf16(
                    false, a.v, false, bfr[mat * 4 + nt].v, (short)0,
                    acc[mat][nt], false, false);
    }

    const int b    = rowBase >> 12;      // / TDIM
    const int tloc = rowBase & (TDIM - 1);
#pragma unroll
    for (int nt = 0; nt < 4; ++nt) {
        const int h = nt * 16 + n16;
        // Q, K: strided 16-bit stores (row varies with v)
#pragma unroll
        for (int v = 0; v < 8; ++v) {
            const int row = rowBase + hl * 8 + v;
            Qw[(size_t)row * HDIM + h] = f32_to_bf16(acc[0][nt][v]);
            Kw[(size_t)row * HDIM + h] = f32_to_bf16(acc[1][nt][v]);
        }
        // V^T: 8 consecutive t per lane -> one 16B store
        uint4 pk;
        pk.x = pack_bf16x2(acc[2][nt][0], acc[2][nt][1]);
        pk.y = pack_bf16x2(acc[2][nt][2], acc[2][nt][3]);
        pk.z = pack_bf16x2(acc[2][nt][4], acc[2][nt][5]);
        pk.w = pack_bf16x2(acc[2][nt][6], acc[2][nt][7]);
        *(uint4*)(Vtw + ((size_t)(b * HDIM + h)) * TDIM + tloc + hl * 8) = pk;
    }
}

// ---------------------------------------------------------------------------
// Async staging: copy one contiguous 4KB K tile (32 rows x 64 cols bf16)
// into LDS with global_load_async_to_lds_b128 (tracked by ASYNCcnt).
// Per lane 16B x 8 issues.  ldsBase = LDS byte offset (low 32 bits of the
// generic pointer: ISA 10.2, LDS aperture keeps the offset in addr[31:0]).
// ---------------------------------------------------------------------------
__device__ __forceinline__ void stage_k_tile(const unsigned short* Kb, int kb,
                                             unsigned int ldsBase, int lane) {
    const unsigned long long sbase = (unsigned long long)Kb;
    const unsigned int g0 = (unsigned int)(kb * HDIM * 2) + lane * 16;
    const unsigned int l0 = ldsBase + lane * 16;
#pragma unroll
    for (int i = 0; i < 8; ++i) {
        unsigned int l = l0 + i * 512;
        unsigned int g = g0 + i * 512;
        asm volatile("global_load_async_to_lds_b128 %0, %1, %2"
                     :: "v"(l), "v"(g), "s"(sbase) : "memory");
    }
}

// ---------------------------------------------------------------------------
// Kernel 2: flash attention.  4 waves/block, each wave = 32 query rows
// (2 C-tiles) so every K/V fragment feeds two WMMAs.  Key tiles of 32.
// K tiles double-buffered in LDS via async loads: tile j+1 is issued before
// consuming tile j (s_wait_asynccnt 8 keeps 8 loads in flight).
// S = Q.K^T (8 WMMA), causal mask only on the diagonal tile, online softmax,
// P transposed via wave-private LDS, O += P.V (8 WMMA).
// Reference semantics: softmax FIRST, then divide by sqrt(H)=8.
// ---------------------------------------------------------------------------
__global__ __launch_bounds__(128) void attn_kernel(
        const unsigned short* __restrict__ Qw,   // [B*T][H] bf16
        const unsigned short* __restrict__ Kw,   // [B*T][H] bf16
        const unsigned short* __restrict__ Vtw,  // [B][H][T] bf16
        float* __restrict__ out) {               // [B*T][H] f32
    __shared__ unsigned short ldsP[4][2][16][32];    // P tiles (8KB)
    __shared__ unsigned short ldsK[4][2][32][64];    // K tiles, 2 bufs (32KB)

    const int lane  = threadIdx.x & 31;
    const int wave  = threadIdx.x >> 5;
    const int hl    = lane >> 4;
    const int n16   = lane & 15;
    const int qbase = blockIdx.x * 128 + wave * 32;  // flat, 32-aligned
    const int b     = qbase >> 12;
    const int qt    = qbase & (TDIM - 1);            // batch-local, 32-aligned

    // Q A-fragments: 2 q-tiles x 2 head chunks, loaded once
    Frag qf[2][2];
#pragma unroll
    for (int qi = 0; qi < 2; ++qi) {
        const unsigned short* qrow =
            Qw + (size_t)(qbase + qi * 16 + n16) * HDIM;
#pragma unroll
        for (int hb = 0; hb < 2; ++hb)
#pragma unroll
            for (int v = 0; v < 8; ++v)
                qf[qi][hb].u[v] =
                    *(const unsigned int*)(qrow + hb * 32 + a_koff(v, hl));
    }

    v8f o[2][4];
    float m_i[2][8], l_i[2][8];
#pragma unroll
    for (int qi = 0; qi < 2; ++qi) {
#pragma unroll
        for (int nt = 0; nt < 4; ++nt) o[qi][nt] = v8f{};
#pragma unroll
        for (int v = 0; v < 8; ++v) {
            m_i[qi][v] = -__builtin_inff();
            l_i[qi][v] = 0.0f;
        }
    }

    const unsigned short* Kb = Kw  + (size_t)b * TDIM * HDIM;
    const unsigned short* Vb = Vtw + (size_t)b * HDIM * TDIM;

    const unsigned int ldsKOff[2] = {
        (unsigned int)(size_t)&ldsK[wave][0][0][0],
        (unsigned int)(size_t)&ldsK[wave][1][0][0]
    };

    const int jlast = qt >> 5;      // both q-tiles share the causal range
    stage_k_tile(Kb, 0, ldsKOff[0], lane);           // prime buffer 0

    for (int j = 0; j <= jlast; ++j) {
        const int kb = j << 5;
        if (j < jlast) {
            stage_k_tile(Kb, kb + 32, ldsKOff[(j + 1) & 1], lane);
            __builtin_prefetch(Vb + (size_t)n16 * TDIM + kb + 32, 0, 1);
            asm volatile("s_wait_asynccnt 0x8" ::: "memory");  // tile j ready
        } else {
            asm volatile("s_wait_asynccnt 0x0" ::: "memory");
        }

        // ---- V fragments early: latency overlaps S WMMA + softmax ----
        Frag vf[4];
#pragma unroll
        for (int nt = 0; nt < 4; ++nt) {
            const unsigned short* vrow =
                Vb + (size_t)(nt * 16 + n16) * TDIM + kb;
#pragma unroll
            for (int v = 0; v < 8; ++v)
                vf[nt].u[v] = *(const unsigned int*)(vrow + hl * 16 + 2 * v);
        }

        // ---- S = Q.K^T : K fragments from the LDS tile ----
        const unsigned short* ktile = &ldsK[wave][j & 1][0][0];
        v8f s[2][2] = { { v8f{}, v8f{} }, { v8f{}, v8f{} } };
#pragma unroll
        for (int t = 0; t < 2; ++t) {
            const unsigned short* krow = ktile + (t * 16 + n16) * 64;
#pragma unroll
            for (int hb = 0; hb < 2; ++hb) {
                Frag kf;
#pragma unroll
                for (int v = 0; v < 8; ++v)
                    kf.u[v] =
                        *(const unsigned int*)(krow + hb * 32 + hl * 16 + 2 * v);
                s[0][t] = __builtin_amdgcn_wmma_f32_16x16x32_bf16(
                    false, qf[0][hb].v, false, kf.v, (short)0, s[0][t],
                    false, false);
                s[1][t] = __builtin_amdgcn_wmma_f32_16x16x32_bf16(
                    false, qf[1][hb].v, false, kf.v, (short)0, s[1][t],
                    false, false);
            }
        }

        // ---- causal mask: only the diagonal tile needs it ----
        if (j == jlast) {
#pragma unroll
            for (int qi = 0; qi < 2; ++qi)
#pragma unroll
                for (int v = 0; v < 8; ++v) {
                    const int row  = qt + qi * 16 + hl * 8 + v;
                    const int key0 = kb + n16;
                    if (key0 > row)      s[qi][0][v] = -__builtin_inff();
                    if (key0 + 16 > row) s[qi][1][v] = -__builtin_inff();
                }
        }

        // ---- online softmax (rows live in 16-lane halves) ----
#pragma unroll
        for (int qi = 0; qi < 2; ++qi) {
#pragma unroll
            for (int v = 0; v < 8; ++v) {
                float mx = fmaxf(s[qi][0][v], s[qi][1][v]);
#pragma unroll
                for (int off = 1; off < 16; off <<= 1)
                    mx = fmaxf(mx, __shfl_xor(mx, off, 32));
                const float mn    = fmaxf(m_i[qi][v], mx);
                const float alpha = __expf(m_i[qi][v] - mn);
                m_i[qi][v] = mn;
                const float p0 = __expf(s[qi][0][v] - mn);
                const float p1 = __expf(s[qi][1][v] - mn);
                float rs = p0 + p1;
#pragma unroll
                for (int off = 1; off < 16; off <<= 1)
                    rs += __shfl_xor(rs, off, 32);
                l_i[qi][v] = l_i[qi][v] * alpha + rs;
#pragma unroll
                for (int nt = 0; nt < 4; ++nt) o[qi][nt][v] *= alpha;
                // C/D layout -> row-major P tile in LDS (bf16)
                const int row = hl * 8 + v;
                ldsP[wave][qi][row][n16]      = f32_to_bf16(p0);
                ldsP[wave][qi][row][16 + n16] = f32_to_bf16(p1);
            }
        }
        // wave-private LDS transpose: order cross-lane store->load
        asm volatile("s_wait_dscnt 0" ::: "memory");

        // P as A-fragments (16x32, K = 32 keys)
        Frag pf[2];
#pragma unroll
        for (int qi = 0; qi < 2; ++qi)
#pragma unroll
            for (int v = 0; v < 8; ++v)
                pf[qi].u[v] =
                    *(const unsigned int*)&ldsP[wave][qi][n16][a_koff(v, hl)];

        // ---- O += P.V : each V fragment feeds both q-tiles ----
#pragma unroll
        for (int nt = 0; nt < 4; ++nt) {
            o[0][nt] = __builtin_amdgcn_wmma_f32_16x16x32_bf16(
                false, pf[0].v, false, vf[nt].v, (short)0, o[0][nt],
                false, false);
            o[1][nt] = __builtin_amdgcn_wmma_f32_16x16x32_bf16(
                false, pf[1].v, false, vf[nt].v, (short)0, o[1][nt],
                false, false);
        }
    }

    // ---- epilogue: out = O / (l * sqrt(H)),  sqrt(H)=8 ----
#pragma unroll
    for (int qi = 0; qi < 2; ++qi)
#pragma unroll
        for (int v = 0; v < 8; ++v) {
            const float inv = 1.0f / (8.0f * l_i[qi][v]);
            const int m = qi * 16 + hl * 8 + v;
#pragma unroll
            for (int nt = 0; nt < 4; ++nt)
                out[(size_t)(qbase + m) * HDIM + nt * 16 + n16] =
                    o[qi][nt][v] * inv;
        }
}

// ---------------------------------------------------------------------------
// Launch
// ---------------------------------------------------------------------------
extern "C" void kernel_launch(void* const* d_in, const int* in_sizes, int n_in,
                              void* d_out, int out_size, void* d_ws, size_t ws_size,
                              hipStream_t stream) {
    const float* x  = (const float*)d_in[0];
    const float* Wq = (const float*)d_in[1];
    const float* Wk = (const float*)d_in[2];
    const float* Wv = (const float*)d_in[3];

    char* ws = (char*)d_ws;
    const size_t wtBytes  = (size_t)3 * HDIM * EDIM * 2;         // 384 KB
    const size_t qkvBytes = (size_t)BDIM * TDIM * HDIM * 2;      // 2 MB each
    unsigned short* Wt  = (unsigned short*)ws;
    unsigned short* Qw  = (unsigned short*)(ws + wtBytes);
    unsigned short* Kw  = (unsigned short*)(ws + wtBytes + qkvBytes);
    unsigned short* Vtw = (unsigned short*)(ws + wtBytes + 2 * qkvBytes);

    // Kernel 0: weight convert + transpose
    {
        const int n = 3 * HDIM * EDIM;
        wprep_kernel<<<(n + 255) / 256, 256, 0, stream>>>(Wq, Wk, Wv, Wt);
    }
    // Kernel 1: QKV projection (bf16 WMMA, f32 accumulate)
    {
        const int rows = BDIM * TDIM;                 // 16384
        qkv_kernel<<<rows / 64, 128, 0, stream>>>(x, Wt, Qw, Kw, Vtw);
    }
    // Kernel 2: causal flash attention, 32 query rows per wave
    {
        const int rows = BDIM * TDIM;
        attn_kernel<<<rows / 128, 128, 0, stream>>>(Qw, Kw, Vtw, (float*)d_out);
    }
}
